// Conv2d_XnorPP_STTN_29119878267347
// MI455X (gfx1250) — 8-bit matrix operand order repaired and checked by reading, compiled, not run on hardware
//
#include <hip/hip_runtime.h>


typedef __attribute__((ext_vector_type(16))) int          v16i;
typedef __attribute__((ext_vector_type(8)))  float        v8f;
typedef __attribute__((ext_vector_type(4)))  unsigned int u32x4;
typedef __attribute__((ext_vector_type(8)))  int          i32x8;
typedef __attribute__((ext_vector_type(4)))  int          i32x4;

#define N_IMG   32
#define C_IN    256
#define C_OUT   256
#define HW      56
#define HP      58
#define KTAP    9
#define KROW    (KTAP * C_IN)

#define XPAD_BYTES   ((size_t)N_IMG * HP * HP * C_IN)
#define WPACK_OFF    XPAD_BYTES
#define WPACK_BYTES  ((size_t)C_OUT * KROW)
#define ALPHA_OFF    (WPACK_OFF + WPACK_BYTES)

__device__ __forceinline__ unsigned char fp8_sign(float v) {
    return v > 0.0f ? 0x38 : (v < 0.0f ? 0xB8 : 0x00);
}

__global__ void zero_xpad_kernel(int4* __restrict__ p) {
    size_t i = (size_t)blockIdx.x * blockDim.x + threadIdx.x;
    int4 z; z.x = 0; z.y = 0; z.z = 0; z.w = 0;
    p[i] = z;
}

__global__ void pack_x_kernel(const float* __restrict__ x,
                              unsigned char* __restrict__ xpad) {
    __shared__ __attribute__((aligned(16))) unsigned char tile[HW * C_IN];
    const int b = blockIdx.x;
    const int n = b / HW, hrow = b % HW;
    const int tid = threadIdx.x;

    for (int idx = tid; idx < HW * C_IN; idx += 256) {
        const int c = idx / HW, w = idx % HW;
        const float v = x[(((size_t)n * C_IN + c) * HW + hrow) * HW + w];
        tile[w * C_IN + c] = fp8_sign(v);
    }
    __syncthreads();

    int* __restrict__ ow = (int*)(xpad + (((size_t)n * HP + (hrow + 1)) * HP + 1) * C_IN);
    const int* __restrict__ tw = (const int*)tile;
    for (int i = tid; i < (HW * C_IN) / 4; i += 256) ow[i] = tw[i];
}

__global__ void pack_w_kernel(const float* __restrict__ w1,
                              const float* __restrict__ w2,
                              unsigned char* __restrict__ wpack,
                              float* __restrict__ alpha) {
    __shared__ float red[256];
    const int o = blockIdx.x, c = threadIdx.x;
    const float* p1 = w1 + ((size_t)o * C_IN + c) * KTAP;
    const float* p2 = w2 + ((size_t)o * C_IN + c) * KTAP;
    float s = 0.0f;
    #pragma unroll
    for (int t = 0; t < KTAP; ++t) {
        const float a = p1[t], b = p2[t];
        s += fabsf(a) + fabsf(b);
        const int wsum = ((a > 0.f) - (a < 0.f)) + ((b > 0.f) - (b < 0.f));
        const int aw = wsum < 0 ? -wsum : wsum;
        unsigned char e = (aw == 0) ? 0x00 : (aw == 1 ? 0x38 : 0x40);
        if (wsum < 0) e |= 0x80;
        wpack[(size_t)o * KROW + t * C_IN + c] = e;
    }
    red[c] = s;
    __syncthreads();
    for (int k = 128; k > 0; k >>= 1) {
        if (c < k) red[c] += red[c + k];
        __syncthreads();
    }
    if (c == 0) alpha[o] = red[0] / (2.0f * (float)KROW);
}

__device__ __forceinline__ void tdm_stage_halo(unsigned lds_base,
                                               unsigned long long gaddr) {
    u32x4 g0;
    g0[0] = 1u;
    g0[1] = lds_base;
    g0[2] = (unsigned)(gaddr & 0xFFFFFFFFull);
    g0[3] = (unsigned)((gaddr >> 32) & 0x01FFFFFFull) | (2u << 30);
    i32x8 g1;
    g1[0] = 3 << 16;
    g1[1] = (int)(0x100000u << 16);
    g1[2] = (int)((0x100000u >> 16) | (0x100000u << 16));
    g1[3] = (int)((0x100000u >> 16) | (320u << 16));
    g1[4] = 6;
    g1[5] = (int)(HP * C_IN / 8);
    g1[6] = 0;
    g1[7] = 0;
    i32x4 gz = {0, 0, 0, 0};
#if defined(__clang_major__) && (__clang_major__ >= 23)
    i32x8 gz8 = {0, 0, 0, 0, 0, 0, 0, 0};
    __builtin_amdgcn_tensor_load_to_lds(g0, g1, gz, gz, gz8, 0);
#else
    __builtin_amdgcn_tensor_load_to_lds(g0, g1, gz, gz, 0);
#endif
}

__global__ __launch_bounds__(256)
void conv_wmma_kernel(const unsigned char* __restrict__ xpad,
                      const unsigned char* __restrict__ wpack,
                      const float* __restrict__ alpha,
                      float* __restrict__ out) {
    __shared__ __attribute__((aligned(16))) unsigned char smem[6 * 10 * C_IN];

    const int bid = blockIdx.x;
    const int n   = bid / 98;
    const int rem = bid % 98;
    const int oy0 = (rem / 7) * 4;
    const int ox0 = (rem % 7) * 8;
    const int tid = threadIdx.x;

    const unsigned char* xbase = xpad + (((size_t)n * HP + oy0) * HP + ox0) * C_IN;
    if (tid < 32) {
        tdm_stage_halo((unsigned)(size_t)&smem[0], (unsigned long long)(size_t)xbase);
        __builtin_amdgcn_s_wait_tensorcnt(0);
    }
    __syncthreads();

    const int wave  = tid >> 5;
    const int lane  = tid & 31;
    const int col   = lane & 15;
    const int h     = lane >> 4;
    const int py    = col >> 2, px = col & 3;
    const int mbase = wave * 32;

    const v8f vzero = {0.f, 0.f, 0.f, 0.f, 0.f, 0.f, 0.f, 0.f};
    v8f acc[2][2] = {{vzero, vzero}, {vzero, vzero}};

    const unsigned char* wlane = wpack + (size_t)(mbase + col) * KROW + (size_t)h * 16;

    for (int tap = 0; tap < KTAP; ++tap) {
        const int dy = tap / 3, dx = tap - dy * 3;
        const unsigned char* bsrc0 = smem + (((py + dy) * 10) + (px + dx)) * C_IN + h * 16;
        const unsigned char* bsrc1 = bsrc0 + 4 * C_IN;
        const unsigned char* arow  = wlane + (size_t)tap * C_IN;
        if (tap < KTAP - 1) {
            __builtin_prefetch(arow + C_IN, 0, 1);
            __builtin_prefetch(arow + C_IN + 16 * KROW, 0, 1);
        }

        #pragma unroll
        for (int kb = 0; kb < C_IN; kb += 128) {
            v16i B0, B1;
            {
                const int4* bp = (const int4*)(bsrc0 + kb);
                const int4 q0 = bp[0], q1 = bp[2], q2 = bp[4], q3 = bp[6];
                B0[0]  = q0.x; B0[1]  = q0.y; B0[2]  = q0.z; B0[3]  = q0.w;
                B0[4]  = q1.x; B0[5]  = q1.y; B0[6]  = q1.z; B0[7]  = q1.w;
                B0[8]  = q2.x; B0[9]  = q2.y; B0[10] = q2.z; B0[11] = q2.w;
                B0[12] = q3.x; B0[13] = q3.y; B0[14] = q3.z; B0[15] = q3.w;
            }
            {
                const int4* bp = (const int4*)(bsrc1 + kb);
                const int4 q0 = bp[0], q1 = bp[2], q2 = bp[4], q3 = bp[6];
                B1[0]  = q0.x; B1[1]  = q0.y; B1[2]  = q0.z; B1[3]  = q0.w;
                B1[4]  = q1.x; B1[5]  = q1.y; B1[6]  = q1.z; B1[7]  = q1.w;
                B1[8]  = q2.x; B1[9]  = q2.y; B1[10] = q2.z; B1[11] = q2.w;
                B1[12] = q3.x; B1[13] = q3.y; B1[14] = q3.z; B1[15] = q3.w;
            }
            #pragma unroll
            for (int t = 0; t < 2; ++t) {
                const int2* ap = (const int2*)(arow + (size_t)t * 16 * KROW + kb);
                const int2 a0 = ap[0],  a1 = ap[1],  a2 = ap[4],  a3 = ap[5];
                const int2 a4 = ap[8],  a5 = ap[9],  a6 = ap[12], a7 = ap[13];
                v16i A;
                A[0]  = a0.x; A[1]  = a0.y; A[2]  = a1.x; A[3]  = a1.y;
                A[4]  = a2.x; A[5]  = a2.y; A[6]  = a3.x; A[7]  = a3.y;
                A[8]  = a4.x; A[9]  = a4.y; A[10] = a5.x; A[11] = a5.y;
                A[12] = a6.x; A[13] = a6.y; A[14] = a7.x; A[15] = a7.y;

                acc[t][0] = __builtin_amdgcn_wmma_f32_16x16x128_fp8_fp8(
                    A, B0, (short)0, acc[t][0], false, false);
                acc[t][1] = __builtin_amdgcn_wmma_f32_16x16x128_fp8_fp8(
                    A, B1, (short)0, acc[t][1], false, false);
            }
        }
    }

    const int oy = oy0 + py;
    float al[2][8];
    #pragma unroll
    for (int t = 0; t < 2; ++t)
        #pragma unroll
        for (int r = 0; r < 8; ++r)
            al[t][r] = alpha[mbase + t * 16 + h * 8 + r];

    #pragma unroll
    for (int t = 0; t < 2; ++t) {
        #pragma unroll
        for (int ng = 0; ng < 2; ++ng) {
            const int ox = ox0 + ng * 4 + px;
            #pragma unroll
            for (int r = 0; r < 8; ++r) {
                const int M = mbase + t * 16 + h * 8 + r;
                out[(((size_t)n * C_OUT + M) * HW + oy) * HW + ox] =
                    acc[t][ng][r] * al[t][r];
            }
        }
    }
}

extern "C" void kernel_launch(void* const* d_in, const int* in_sizes, int n_in,
                              void* d_out, int out_size, void* d_ws, size_t ws_size,
                              hipStream_t stream) {
    (void)in_sizes; (void)n_in; (void)out_size; (void)ws_size;
    const float* x  = (const float*)d_in[0];
    const float* w1 = (const float*)d_in[1];
    const float* w2 = (const float*)d_in[2];
    float* out = (float*)d_out;

    unsigned char* ws    = (unsigned char*)d_ws;
    unsigned char* xpad  = ws;
    unsigned char* wpack = ws + WPACK_OFF;
    float*         alpha = (float*)(ws + ALPHA_OFF);

    zero_xpad_kernel<<<6728, 256, 0, stream>>>((int4*)xpad);
    pack_x_kernel<<<N_IMG * HW, 256, 0, stream>>>(x, xpad);
    pack_w_kernel<<<C_OUT, 256, 0, stream>>>(w1, w2, wpack, alpha);
    conv_wmma_kernel<<<3136, 256, 0, stream>>>(xpad, wpack, alpha, out);
}
